// Decoder_85942295593401
// MI455X (gfx1250) — matrix operand order repaired and checked by reading, not compiled, not run on hardware
//
#include <hip/hip_runtime.h>

// Problem constants (from reference): B, T, NVAR, K, E
#define BB   4
#define TT   2048
#define NV   128
#define KT   16
#define EE   512
#define TM1  2047   // T-1 output rows per batch

typedef __attribute__((ext_vector_type(16))) _Float16 v16h;
typedef __attribute__((ext_vector_type(8)))  _Float16 v8h;
typedef __attribute__((ext_vector_type(8)))  float    v8f;

// workspace layout (bytes)
#define WT_BYTES  (NV * NV * KT * 2)          // 524288: Wt[o][k][i] f16
#define CNT_OFF   (WT_BYTES)                  // 128 ints
#define LST_OFF   (WT_BYTES + 512)            // 128*512 ints
// total ~787 KB

// ---------------------------------------------------------------------------
// Prep 1: conv_weight (o,i,k) f32  ->  Wt[o][k][i] f16, diagonal last-tap = 0
// ---------------------------------------------------------------------------
__global__ void prep_w_kernel(const float* __restrict__ w,
                              _Float16* __restrict__ wt) {
  int idx = blockIdx.x * blockDim.x + threadIdx.x;   // over NV*KT*NV
  if (idx >= NV * NV * KT) return;
  int i = idx & (NV - 1);          // fastest in output
  int k = (idx >> 7) & (KT - 1);
  int o = idx >> 11;
  float v = w[(o * NV + i) * KT + k];
  if (o == i && k == KT - 1) v = 0.f;
  wt[idx] = (_Float16)v;
}

// ---------------------------------------------------------------------------
// Prep 2: deterministic per-receiver edge lists (CSR with fixed capacity EE)
// ---------------------------------------------------------------------------
__global__ void prep_edges_kernel(const int* __restrict__ recv,
                                  int* __restrict__ counts,
                                  int* __restrict__ lists) {
  int o = threadIdx.x;
  if (o >= NV) return;
  int c = 0;
  for (int e = 0; e < EE; ++e)
    if (recv[e] == o) lists[o * EE + (c++)] = e;
  counts[o] = c;
}

// ---------------------------------------------------------------------------
// Main conv as WMMA GEMM.
//   out[b,tau,o] = bias[o] + sum_{k,i} X[b,i,tau+k-14] * W[o,i,k]
// Inner dim kappa = k*128 + i (tap-major). Per WMMA step c (64 steps):
//   k = c>>2 fixed, i-block = 32*(c&3).
// Workgroup: 256 threads = 8 waves, tile M=64 (tau) x N=128 (o).
// Wave (2x4 grid) computes a 32x32 tile: 2 A-frags, 2 B-frags, 4 accums.
// LDS: Xt[u][i] time-major f16, u in [0,80), row stride 136 (bank spread).
// ---------------------------------------------------------------------------
#define XS 136

__global__ __launch_bounds__(256) void conv_wmma_kernel(
    const float* __restrict__ spikes,     // (B,T,NV)
    const _Float16* __restrict__ wt,      // Wt[o][k][i]
    const float* __restrict__ bias,       // (NV)
    float* __restrict__ out) {            // (B,TM1,NV)
  __shared__ __attribute__((aligned(16))) _Float16 Xt[80 * XS];

  const int b    = blockIdx.x >> 5;
  const int mb   = blockIdx.x & 31;
  const int tau0 = mb * 64;
  const int tid  = threadIdx.x;

  // Stage time window t in [tau0-14, tau0+65] as f16, zero-padded OOB.
  for (int it = 0; it < 40; ++it) {
    int idx = it * 256 + tid;
    int u = idx >> 7, i = idx & 127;          // i fastest -> coalesced reads
    int t = tau0 - 14 + u;
    float v = (t >= 0 && t < TT) ? spikes[(b * TT + t) * NV + i] : 0.f;
    Xt[u * XS + i] = (_Float16)v;
  }
  __syncthreads();

  const int wave = tid >> 5;
  const int lane = tid & 31;
  const int Mw = (wave >> 2) * 32;   // 0 or 32   (tau offset)
  const int Nw = (wave & 3) * 32;    // 0..96     (o offset)
  const int m  = lane & 15;
  const int h  = lane >> 4;

  v8f acc[2][2];
  #pragma unroll
  for (int mi = 0; mi < 2; ++mi)
    #pragma unroll
    for (int ni = 0; ni < 2; ++ni)
      acc[mi][ni] = (v8f){0.f, 0.f, 0.f, 0.f, 0.f, 0.f, 0.f, 0.f};

  for (int c = 0; c < 64; ++c) {
    const int k  = c >> 2;
    const int ia = (c & 3) * 32 + 16 * h;  // A-frag var base (mult of 16)
    const int ib = (c & 3) * 32 + 16 * h;  // B-frag var base (mult of 16)

    // A fragments: lane holds row tau = base+m at fixed time u = base+m+k;
    // elements 0..15 = vars ia..ia+15.
    v16h a[2];
    #pragma unroll
    for (int mi = 0; mi < 2; ++mi) {
      int u = Mw + mi * 16 + m + k;
      const v8h* p = reinterpret_cast<const v8h*>(&Xt[u * XS + ia]);
      v8h lo = p[0];
      v8h hi = p[1];                       // +8 f16 elements
      a[mi] = __builtin_shufflevector(lo, hi,
                0, 1, 2, 3, 4, 5, 6, 7, 8, 9, 10, 11, 12, 13, 14, 15);
    }

    // B fragments: lane column o = Nw+ni*16+m, elements = 16 contiguous vars.
    #pragma unroll
    for (int ni = 0; ni < 2; ++ni) {
      int o = Nw + ni * 16 + m;
      v16h bf = *reinterpret_cast<const v16h*>(&wt[(o * KT + k) * NV + ib]);
      #pragma unroll
      for (int mi = 0; mi < 2; ++mi)
        acc[mi][ni] = __builtin_amdgcn_wmma_f32_16x16x32_f16(
            false, a[mi], false, bf, (short)0, acc[mi][ni], false, false);
    }
  }

  // Epilogue: C/D layout -> VGPR r: lanes 0-15 row r, lanes 16-31 row r+8.
  #pragma unroll
  for (int ni = 0; ni < 2; ++ni) {
    int o = Nw + ni * 16 + m;
    float bv = bias[o];
    #pragma unroll
    for (int mi = 0; mi < 2; ++mi) {
      #pragma unroll
      for (int r = 0; r < 8; ++r) {
        int tau = tau0 + Mw + mi * 16 + r + 8 * h;
        if (tau < TM1)
          out[(b * TM1 + tau) * NV + o] = acc[mi][ni][r] + bv;
      }
    }
  }
}

// ---------------------------------------------------------------------------
// Dynamic-edge streaming kernel: the bandwidth-dominant part (268 MB dynW).
//   out[b,tau,o] += sum_{e in recv-list[o]} sum_k X[b,send[e],tau+k-14]*dynW
// Spikes are exactly 0/1 -> keep the X window as a BITMASK in LDS:
//   Xb[v][w] bit u  =  (X[b,v,tau0-14+u] != 0),  u in [0,144).
// Per edge a thread derives the 16-tap mask with 2 ds_load_b32 + 64-bit
// shift; if the mask is zero (P = 0.9^16 ~ 18.5%) the four 16-byte dynW
// loads are skipped under EXEC -> ~18% less HBM traffic on the hot stream.
// Accumulation = exact fp32 sum of selected weights (bit-exact vs x*w).
// Each (b,tau,o) is owned by exactly one thread -> deterministic, no atomics.
// ---------------------------------------------------------------------------
#define DW   6    // u32 words per var (covers u in [0,192))

__global__ __launch_bounds__(128) void dyn_kernel(
    const float* __restrict__ spikes,
    const float* __restrict__ dynw,       // (E,B,TM1,KT)
    const int* __restrict__ send,
    const int* __restrict__ counts,
    const int* __restrict__ lists,
    float* __restrict__ out) {
  __shared__ unsigned int Xb[NV * DW];    // 3 KB bitmask

  const int oc   = blockIdx.x & 3;        // o-chunk of 32
  const int tb   = (blockIdx.x >> 2) & 15;
  const int b    = blockIdx.x >> 6;
  const int tau0 = tb * 128;
  const int tid  = threadIdx.x;

  // zero bitmask
  #pragma unroll
  for (int w = 0; w < DW; ++w) Xb[w * 128 + tid] = 0u;
  __syncthreads();

  // build bitmask: window u in [0,144) -> t = tau0-14+u (coalesced reads)
  for (int it = 0; it < 144; ++it) {
    int idx = it * 128 + tid;
    int u = idx >> 7, i = idx & 127;
    int t = tau0 - 14 + u;
    float v = (t >= 0 && t < TT) ? spikes[(b * TT + t) * NV + i] : 0.f;
    if (v != 0.f)
      atomicOr(&Xb[i * DW + (u >> 5)], 1u << (u & 31));
  }
  __syncthreads();

  const int tau = tau0 + tid;
  if (tau >= TM1) return;

  const int w0 = tid >> 5;                // per-thread constant word base
  const int sh = tid & 31;                // per-thread constant bit shift

  for (int o = oc * 32; o < oc * 32 + 32; ++o) {
    float accum = 0.f;
    const int cnt = counts[o];
    const int* lst = &lists[o * EE];
    for (int j = 0; j < cnt; ++j) {
      int e = lst[j];
      int v = send[e];
      // 16-tap spike mask for taps k=0..15 (bits u = tid+k)
      unsigned int lo = Xb[v * DW + w0];
      unsigned int hi = Xb[v * DW + w0 + 1];
      unsigned long long comb =
          ((unsigned long long)hi << 32) | (unsigned long long)lo;
      unsigned int msk = (unsigned int)(comb >> sh) & 0xffffu;

      if (j + 1 < cnt) {  // prefetch next edge's dynW line
        int en = lst[j + 1];
        __builtin_prefetch(&dynw[((en * BB + b) * TM1 + tau) * KT], 0, 1);
      }

      if (msk) {
        const float4* dw4 = reinterpret_cast<const float4*>(
            &dynw[((e * BB + b) * TM1 + tau) * KT]);   // 64B aligned
        float4 d0 = dw4[0], d1 = dw4[1], d2 = dw4[2], d3 = dw4[3];
        float s = 0.f;
        s += (msk & 0x0001u) ? d0.x : 0.f;
        s += (msk & 0x0002u) ? d0.y : 0.f;
        s += (msk & 0x0004u) ? d0.z : 0.f;
        s += (msk & 0x0008u) ? d0.w : 0.f;
        s += (msk & 0x0010u) ? d1.x : 0.f;
        s += (msk & 0x0020u) ? d1.y : 0.f;
        s += (msk & 0x0040u) ? d1.z : 0.f;
        s += (msk & 0x0080u) ? d1.w : 0.f;
        s += (msk & 0x0100u) ? d2.x : 0.f;
        s += (msk & 0x0200u) ? d2.y : 0.f;
        s += (msk & 0x0400u) ? d2.z : 0.f;
        s += (msk & 0x0800u) ? d2.w : 0.f;
        s += (msk & 0x1000u) ? d3.x : 0.f;
        s += (msk & 0x2000u) ? d3.y : 0.f;
        s += (msk & 0x4000u) ? d3.z : 0.f;
        s += (msk & 0x8000u) ? d3.w : 0.f;
        accum += s;
      }
    }
    out[(b * TM1 + tau) * NV + o] += accum;
  }
}

// ---------------------------------------------------------------------------
extern "C" void kernel_launch(void* const* d_in, const int* in_sizes, int n_in,
                              void* d_out, int out_size, void* d_ws, size_t ws_size,
                              hipStream_t stream) {
  (void)in_sizes; (void)n_in; (void)out_size; (void)ws_size;

  const float* spikes = (const float*)d_in[0];   // (B,T,NV,1)
  const float* convw  = (const float*)d_in[1];   // (NV,NV,KT)
  const float* bias   = (const float*)d_in[2];   // (NV)
  const float* dynw   = (const float*)d_in[3];   // (E,B,T-1,KT)
  const int*   send   = (const int*)d_in[4];     // (E)
  const int*   recv   = (const int*)d_in[5];     // (E)
  float*       out    = (float*)d_out;           // (B,T-1,NV,1)

  char* ws = (char*)d_ws;
  _Float16* wt  = (_Float16*)ws;
  int* counts   = (int*)(ws + CNT_OFF);
  int* lists    = (int*)(ws + LST_OFF);

  // 1) weight transpose/convert (f32 -> f16, [o][k][i], diag tap zeroed)
  prep_w_kernel<<<(NV * NV * KT + 255) / 256, 256, 0, stream>>>(convw, wt);

  // 2) deterministic receiver edge lists
  prep_edges_kernel<<<1, 128, 0, stream>>>(recv, counts, lists);

  // 3) conv via WMMA: B*32 workgroups of 256 threads (tile 64 tau x 128 o)
  conv_wmma_kernel<<<BB * 32, 256, 0, stream>>>(spikes, wt, bias, out);

  // 4) dynamic messages: stream dyn_weights (skipping all-zero windows)
  dyn_kernel<<<BB * 16 * 4, 128, 0, stream>>>(spikes, dynw, send, counts,
                                              lists, out);
}
